// Encoder_16612933500999
// MI455X (gfx1250) — matrix operand order repaired and checked by reading, not compiled, not run on hardware
//
#include <hip/hip_runtime.h>
#include <hip/hip_bf16.h>

// ---------------------------------------------------------------------------
// MI455X / gfx1250: bf16 WMMA GEMMs with Tensor-Data-Mover (TDM) LDS staging.
//  pre-pass:  x -> bf16;  W_enc -> (e,d,f) bf16;  W_dec -> (e,f,d) bf16
//  K1: out = x @ W_enc + b_enc      (TDM double-buffered, v_wmma bf16)
//  K2: decoder GEMM fused with complex-corr scoring (TDM double-buffered)
//  K3: argmax over experts + gather
// ---------------------------------------------------------------------------

typedef __attribute__((ext_vector_type(16))) __bf16 v16bf;
typedef __attribute__((ext_vector_type(8)))  __bf16 v8bf;
typedef __attribute__((ext_vector_type(8)))  float  v8f;
typedef __attribute__((ext_vector_type(4)))  float  v4f;
typedef __attribute__((ext_vector_type(4)))  unsigned int u32x4;
typedef __attribute__((ext_vector_type(8)))  int    i32x8;
typedef __attribute__((ext_vector_type(4)))  int    i32x4;

#define E_    5
#define B_    4096
#define F_    9984
#define D_    512
#define NSUB_ 13
#define SUBL_ 768   // F_/NSUB_

#define SA 40    // LDS row stride (bf16) for 32-wide K tiles  (32 + 8 pad)
#define SB 520   // LDS row stride (bf16) for 512-deep K tiles (512 + 8 pad)

__device__ __forceinline__ v16bf concat8(v8bf lo, v8bf hi) {
    return __builtin_shufflevector(lo, hi, 0,1,2,3,4,5,6,7,8,9,10,11,12,13,14,15);
}

__device__ __forceinline__ unsigned int lds_addr_of(const void* p) {
    // generic LDS address = {aperture_hi32, wave-relative LDS offset}
    return (unsigned int)(uintptr_t)p;
}

// 2-D TDM load: tile_d1 rows of tile_d0 bf16 elements, global row stride
// stride0 (elements), written contiguously to LDS with pad_amount(code)
// dwords inserted every pad_interval(code) chunks (ISA 8.4).
__device__ __forceinline__ void tdm_load_2d(const __bf16* gsrc, unsigned int lds,
                                            unsigned int tile_d0, unsigned int tile_d1,
                                            unsigned long long stride0,
                                            unsigned int pad_int, unsigned int pad_amt)
{
    const unsigned long long ga = (unsigned long long)(uintptr_t)gsrc;
    u32x4 g0;
    g0[0] = 1u;                                               // count=1 (valid), user mode
    g0[1] = lds;                                              // lds_addr (bytes)
    g0[2] = (unsigned int)ga;                                 // global_addr[31:0]
    g0[3] = (unsigned int)((ga >> 32) & 0x1FFFFFFull) | (2u << 30);  // addr[56:32], type=2
    i32x8 g1;
    g1[0] = (int)((1u << 16) |                                // data_size = 2 bytes
                  (1u << 20) |                                // pad_enable
                  (pad_int << 22) | (pad_amt << 25));
    const unsigned long long td0 = stride0;                   // tensor_dim0 (row length)
    const unsigned int       td1 = tile_d1;                   // tensor_dim1
    g1[1] = (int)((td0 & 0xFFFFu) << 16);                     // [63:48] = tensor_dim0 lo
    g1[2] = (int)(((td0 >> 16) & 0xFFFFu) | ((td1 & 0xFFFFu) << 16));
    g1[3] = (int)(((td1 >> 16) & 0xFFFFu) | (tile_d0 << 16)); // [127:112] tile_dim0
    g1[4] = (int)(tile_d1 & 0xFFFFu);                         // tile_dim1 (tile_dim2=0)
    g1[5] = (int)(unsigned int)(stride0 & 0xFFFFFFFFull);     // tensor_dim0_stride lo
    g1[6] = (int)(unsigned int)((stride0 >> 32) & 0xFFFFull); // stride hi (dim1 stride=0)
    g1[7] = 0;
    i32x4 z4; z4[0] = 0; z4[1] = 0; z4[2] = 0; z4[3] = 0;
#if __has_include(<hip/amd_detail/amd_gfx1250_TDM.h>)
    i32x8 z8; for (int q = 0; q < 8; ++q) z8[q] = 0;
    __builtin_amdgcn_tensor_load_to_lds(g0, g1, z4, z4, z8, 0);
#else
    __builtin_amdgcn_tensor_load_to_lds(g0, g1, z4, z4, 0);
#endif
}

// ---------------------------------------------------------------------------
// Pre-pass: f32 -> bf16 elementwise
// ---------------------------------------------------------------------------
__global__ __launch_bounds__(256)
void cvt_bf16_kernel(const float* __restrict__ s, __bf16* __restrict__ d, int n4)
{
    const int i = (blockIdx.x * 256 + threadIdx.x);
    if (i < n4) {
        v4f v = *(const v4f*)(s + (size_t)i * 4);
        __bf16* o = d + (size_t)i * 4;
        o[0] = (__bf16)v.x; o[1] = (__bf16)v.y; o[2] = (__bf16)v.z; o[3] = (__bf16)v.w;
    }
}

// ---------------------------------------------------------------------------
// Pre-pass: per-expert transpose + convert.  src slice (R x C) f32 -> dst (C x R) bf16
// ---------------------------------------------------------------------------
__global__ __launch_bounds__(256)
void transpose_cvt_kernel(const float* __restrict__ src, __bf16* __restrict__ dst,
                          int R, int C)
{
    __shared__ __bf16 t[32][33];
    const int e  = blockIdx.z;
    src += (size_t)e * R * C;
    dst += (size_t)e * R * C;
    const int c0 = blockIdx.x * 32;
    const int r0 = blockIdx.y * 32;
    const int tx = threadIdx.x & 31;
    const int ty = threadIdx.x >> 5;          // 0..7
    for (int q = 0; q < 4; ++q) {
        const int r = ty + q * 8;
        t[r][tx] = (__bf16)src[(size_t)(r0 + r) * C + c0 + tx];
    }
    __syncthreads();
    for (int q = 0; q < 4; ++q) {
        const int r = ty + q * 8;             // dst row = original column
        dst[(size_t)(c0 + r) * R + r0 + tx] = t[tx][r];
    }
}

// ---------------------------------------------------------------------------
// Kernel 1: out[e,b,d] = x @ W_enc[e] + b_enc[e]   (f32 + bf16 copies out)
// Block tile 128x128, BK=32, TDM double-buffered LDS, 8 waves x (4x2) WMMA.
// xh: (B,F) bf16 ;  Wet: (E,D,F) bf16 (transposed W_enc)
// ---------------------------------------------------------------------------
__global__ __launch_bounds__(256)
void enc_gemm_kernel(const __bf16* __restrict__ xh, const __bf16* __restrict__ Wet,
                     const float* __restrict__ be,
                     float* __restrict__ outF, __bf16* __restrict__ outH)
{
    __shared__ __bf16 As[2][128 * SA];   // x tile,     As[m][k]
    __shared__ __bf16 Bs[2][128 * SA];   // W_enc tile, Bs[n][k]

    const int m0   = blockIdx.x * 128;
    const int n0   = blockIdx.y * 128;
    const int e    = blockIdx.z;
    const int tid  = threadIdx.x;
    const int lane = tid & 31;
    const int wave = tid >> 5;
    const int wm   = wave >> 2;   // 0..1
    const int wn   = wave & 3;    // 0..3
    const int l16  = lane & 15;
    const int lh   = lane >> 4;

    const __bf16* Abase = xh + (size_t)m0 * F_;
    const __bf16* Bbase = Wet + ((size_t)e * D_ + n0) * F_;

    v8f acc[4][2];
    for (int i = 0; i < 4; ++i)
        for (int j = 0; j < 2; ++j)
            for (int q = 0; q < 8; ++q) acc[i][j][q] = 0.0f;

    if (wave == 0) {
        tdm_load_2d(Abase, lds_addr_of(&As[0][0]), 32, 128, F_, 3, 3);
        tdm_load_2d(Bbase, lds_addr_of(&Bs[0][0]), 32, 128, F_, 3, 3);
    }

    const int NK = F_ / 32;   // 312
    for (int ki = 0; ki < NK; ++ki) {
        const int cur = ki & 1;
        if (wave == 0) {
            if (ki + 1 < NK) {
                tdm_load_2d(Abase + (size_t)(ki + 1) * 32,
                            lds_addr_of(&As[cur ^ 1][0]), 32, 128, F_, 3, 3);
                tdm_load_2d(Bbase + (size_t)(ki + 1) * 32,
                            lds_addr_of(&Bs[cur ^ 1][0]), 32, 128, F_, 3, 3);
                __builtin_amdgcn_s_wait_tensorcnt(2);   // current tile landed
            } else {
                __builtin_amdgcn_s_wait_tensorcnt(0);
            }
        }
        __syncthreads();

        v16bf af[4], bfr[2];
        for (int i = 0; i < 4; ++i) {
            const int m = wm * 64 + i * 16 + l16;
            v8bf lo = *(const v8bf*)&As[cur][m * SA + lh * 8];
            v8bf hi = *(const v8bf*)&As[cur][m * SA + lh * 8 + 16];
            af[i] = concat8(lo, hi);
        }
        for (int j = 0; j < 2; ++j) {
            const int n = wn * 32 + j * 16 + l16;
            v8bf lo = *(const v8bf*)&Bs[cur][n * SA + lh * 8];
            v8bf hi = *(const v8bf*)&Bs[cur][n * SA + lh * 8 + 16];
            bfr[j] = concat8(lo, hi);
        }
        for (int i = 0; i < 4; ++i)
            for (int j = 0; j < 2; ++j)
                acc[i][j] = __builtin_amdgcn_wmma_f32_16x16x32_bf16(
                    false, af[i], false, bfr[j], (short)0, acc[i][j], false, false);
        __syncthreads();   // readers done before TDM overwrites this buffer
    }

    for (int i = 0; i < 4; ++i)
        for (int j = 0; j < 2; ++j) {
            const int n    = n0 + wn * 32 + j * 16 + l16;
            const float bv = be[e * D_ + n];
            const int mb   = m0 + wm * 64 + i * 16 + lh * 8;
            for (int r = 0; r < 8; ++r) {
                const float v  = acc[i][j][r] + bv;
                const size_t o = ((size_t)e * B_ + (mb + r)) * D_ + n;
                outF[o] = v;
                outH[o] = (__bf16)v;
            }
        }
}

// ---------------------------------------------------------------------------
// Kernel 2: decoder GEMM fused with complex-corr scoring.
// One block per (e, sub, 128-row strip).  Wave w owns rows [16w,16w+16);
// A strip (16x512 bf16) in 16 register fragments.  24 chunks of 32 f-cols,
// TDM double-buffered W_dec staging; y folded into i/j partials in-register.
// Wdt: (E,F,D) bf16 (transposed W_dec)
// ---------------------------------------------------------------------------
__global__ __launch_bounds__(256)
void dec_score_kernel(const __bf16* __restrict__ outH, const __bf16* __restrict__ Wdt,
                      const float* __restrict__ bd, const float* __restrict__ x,
                      float* __restrict__ scores)
{
    __shared__ __bf16 Bs[2][32 * SB];    // W_dec chunk, Bs[n][k]
    __shared__ float  lds_i[128];
    __shared__ float  lds_j[128];

    const int e    = blockIdx.z;
    const int sub  = blockIdx.y;
    const int b0   = blockIdx.x * 128;
    const int tid  = threadIdx.x;
    const int lane = tid & 31;
    const int wave = tid >> 5;
    const int l16  = lane & 15;
    const int lh   = lane >> 4;

    if (tid < 128) { lds_i[tid] = 0.0f; lds_j[tid] = 0.0f; }

    const int fbase = sub * SUBL_;
    const __bf16* Wbase = Wdt + ((size_t)e * F_ + fbase) * D_;

    if (wave == 0)
        tdm_load_2d(Wbase, lds_addr_of(&Bs[0][0]), 512, 32, D_, 7, 3);

    // A strip into registers: 16 fragments covering K=0..511
    const int mrow = b0 + wave * 16 + l16;
    const __bf16* Arow = outH + ((size_t)e * B_ + mrow) * D_;
    v16bf af[16];
    for (int kf = 0; kf < 16; ++kf) {
        v8bf lo = *(const v8bf*)(Arow + kf * 32 + lh * 8);
        v8bf hi = *(const v8bf*)(Arow + kf * 32 + lh * 8 + 16);
        af[kf] = concat8(lo, hi);
    }

    float pi[8], pj[8];
    for (int r = 0; r < 8; ++r) { pi[r] = 0.0f; pj[r] = 0.0f; }

    const int NCH = SUBL_ / 32;   // 24
    for (int ch = 0; ch < NCH; ++ch) {
        const int cur = ch & 1;
        if (wave == 0) {
            if (ch + 1 < NCH) {
                tdm_load_2d(Wbase + (size_t)(ch + 1) * 32 * D_,
                            lds_addr_of(&Bs[cur ^ 1][0]), 512, 32, D_, 7, 3);
                __builtin_amdgcn_s_wait_tensorcnt(1);
            } else {
                __builtin_amdgcn_s_wait_tensorcnt(0);
            }
        }
        __syncthreads();

        v8f c0, c1;
        for (int q = 0; q < 8; ++q) { c0[q] = 0.0f; c1[q] = 0.0f; }
        for (int kf = 0; kf < 16; ++kf) {
            const v8bf* p0 = (const v8bf*)&Bs[cur][l16 * SB + kf * 32 + lh * 8];
            const v8bf* p1 = (const v8bf*)&Bs[cur][(16 + l16) * SB + kf * 32 + lh * 8];
            v16bf b0f = concat8(p0[0], p0[2]);
            v16bf b1f = concat8(p1[0], p1[2]);
            c0 = __builtin_amdgcn_wmma_f32_16x16x32_bf16(
                false, af[kf], false, b0f, (short)0, c0, false, false);
            c1 = __builtin_amdgcn_wmma_f32_16x16x32_bf16(
                false, af[kf], false, b1f, (short)0, c1, false, false);
        }
        __syncthreads();   // readers done before TDM overwrites this buffer

        //   i += y[m,f]*x[m,f];  j += y[m,f] * (f even ? x[m,f+1] : -x[m,f-1])
        const int f0   = fbase + ch * 32 + l16;
        const float sg = (l16 & 1) ? -1.0f : 1.0f;
        for (int half = 0; half < 2; ++half) {
            const int f  = f0 + half * 16;
            const int fp = f ^ 1;
            const v8f& c = half ? c1 : c0;
            for (int r = 0; r < 8; ++r) {
                const float* xr = x + (size_t)(b0 + wave * 16 + r + 8 * lh) * F_;
                const float  y  = c[r];
                pi[r] += y * xr[f];
                pj[r] += sg * y * xr[fp];
            }
        }
    }

    for (int r = 0; r < 8; ++r) {
        const int ml = wave * 16 + r + 8 * lh;
        atomicAdd(&lds_i[ml], pi[r]);
        atomicAdd(&lds_j[ml], pj[r]);
    }
    {   // b_dec contribution
        const int row = tid & 127;
        const int h   = tid >> 7;
        const float* xr = x  + (size_t)(b0 + row) * F_ + fbase;
        const float* br = bd + (size_t)e * F_ + fbase;
        float bi = 0.0f, bj = 0.0f;
        for (int p = h * 192; p < (h + 1) * 192; ++p) {
            float2 xv = *(const float2*)(xr + 2 * p);
            float2 bv = *(const float2*)(br + 2 * p);
            bi += bv.x * xv.x + bv.y * xv.y;
            bj += bv.x * xv.y - bv.y * xv.x;
        }
        atomicAdd(&lds_i[row], bi);
        atomicAdd(&lds_j[row], bj);
    }
    __syncthreads();
    if (tid < 128) {
        const float iv = lds_i[tid], jv = lds_j[tid];
        atomicAdd(&scores[(size_t)e * B_ + b0 + tid],
                  (iv * iv + jv * jv) * (1.0f / (float)NSUB_));
    }
}

// ---------------------------------------------------------------------------
__global__ void zero_scores_kernel(float* __restrict__ s)
{
    const int i = blockIdx.x * 256 + threadIdx.x;
    if (i < E_ * B_) s[i] = 0.0f;
}

__global__ __launch_bounds__(256)
void argmax_gather_kernel(const float* __restrict__ scores,
                          const float* __restrict__ outF,
                          float* __restrict__ sel)
{
    const int wave = threadIdx.x >> 5;
    const int lane = threadIdx.x & 31;
    const int b    = blockIdx.x * 8 + wave;

    float best = scores[b];
    int   bi   = 0;
    for (int e = 1; e < E_; ++e) {
        const float s = scores[(size_t)e * B_ + b];
        if (s > best) { best = s; bi = e; }
    }
    const float* src = outF + ((size_t)bi * B_ + b) * D_;
    float*       dst = sel + (size_t)b * D_;
    for (int q = 0; q < 4; ++q) {
        const int off = q * 128 + lane * 4;
        *(v4f*)(dst + off) = *(const v4f*)(src + off);
    }
}

// ---------------------------------------------------------------------------
extern "C" void kernel_launch(void* const* d_in, const int* in_sizes, int n_in,
                              void* d_out, int out_size, void* d_ws, size_t ws_size,
                              hipStream_t stream)
{
    (void)in_sizes; (void)n_in; (void)out_size; (void)ws_size;

    const float* x  = (const float*)d_in[0];
    const float* We = (const float*)d_in[1];
    const float* be = (const float*)d_in[2];
    const float* Wd = (const float*)d_in[3];
    const float* bd = (const float*)d_in[4];
    float*       sel = (float*)d_out;

    // workspace layout (256B-aligned):
    //   xh     : B*F   bf16  (81.8 MB)
    //   Wet    : E*D*F bf16  (51.1 MB)   W_enc transposed (e,d,f)
    //   Wdt    : E*F*D bf16  (51.1 MB)   W_dec transposed (e,f,d)
    //   outF   : E*B*D f32   (41.9 MB)
    //   outH   : E*B*D bf16  (21.0 MB)
    //   scores : E*B   f32
    char* ws = (char*)d_ws;
    const size_t nX   = (size_t)B_ * F_;
    const size_t nW   = (size_t)E_ * F_ * D_;
    const size_t nOut = (size_t)E_ * B_ * D_;
    #define ALIGN256(v) (((v) + 255) & ~(size_t)255)
    __bf16* xh     = (__bf16*)ws;
    __bf16* Wet    = (__bf16*)(ws + ALIGN256(nX * 2));
    __bf16* Wdt    = (__bf16*)((char*)Wet + ALIGN256(nW * 2));
    float*  outF   = (float*)((char*)Wdt + ALIGN256(nW * 2));
    __bf16* outH   = (__bf16*)((char*)outF + ALIGN256(nOut * 4));
    float*  scores = (float*)((char*)outH + ALIGN256(nOut * 2));

    zero_scores_kernel<<<(E_ * B_ + 255) / 256, 256, 0, stream>>>(scores);

    // pre-pass conversions
    cvt_bf16_kernel<<<(int)(nX / 4 + 255) / 256, 256, 0, stream>>>(x, xh, (int)(nX / 4));
    dim3 gt1(D_ / 32, F_ / 32, E_);   // W_enc (F x D) -> (D x F)
    transpose_cvt_kernel<<<gt1, 256, 0, stream>>>(We, Wet, F_, D_);
    dim3 gt2(F_ / 32, D_ / 32, E_);   // W_dec (D x F) -> (F x D)
    transpose_cvt_kernel<<<gt2, 256, 0, stream>>>(Wd, Wdt, D_, F_);

    dim3 g1(B_ / 128, D_ / 128, E_);  // 32 x 4 x 5
    enc_gemm_kernel<<<g1, 256, 0, stream>>>(xh, Wet, be, outF, outH);

    dim3 g2(B_ / 128, NSUB_, E_);     // 32 x 13 x 5
    dec_score_kernel<<<g2, 256, 0, stream>>>(outH, Wdt, bd, x, scores);

    argmax_gather_kernel<<<B_ / 8, 256, 0, stream>>>(scores, outF, sel);
}
